// GATv2Layer_66632122630755
// MI455X (gfx1250) — matrix operand order repaired and checked by reading, not compiled, not run on hardware
//
#include <hip/hip_runtime.h>
#include <hip/hip_bf16.h>

typedef __attribute__((ext_vector_type(16))) _Float16 v16h;
typedef __attribute__((ext_vector_type(8)))  _Float16 v8h;
typedef __attribute__((ext_vector_type(8)))  float    v8f;

#define BQ 8
#define NN 1024
#define HH 8
#define DD 32
#define CIN 256
#define COUT 256

// ---------------------------------------------------------------------------
// Kernel 1: h = x @ W^T   (M = B*N = 8192, N = 256, K = 256)
// Each wave computes a 16x64 tile: one A frag (f16 16x32) reused across 4 B frags.
// Writes h as fp32 [8192,256] and transposed f16 hT[b][h][d][n].
// ---------------------------------------------------------------------------
__global__ __launch_bounds__(256) void gat_gemm1(
    const float* __restrict__ x, const float* __restrict__ W,
    float* __restrict__ hf32, _Float16* __restrict__ hT) {
  const int wave = blockIdx.x * 8 + (threadIdx.x >> 5);
  const int lane = threadIdx.x & 31;
  const int mt = wave >> 2;        // 0..511  (16-row tile of 8192 rows)
  const int nq = wave & 3;         // 0..3    (64-col quad of 256 cols)
  const int row0 = mt * 16;
  const int o0 = nq * 64;

  const int m = lane & 15;         // A row / B col within tile
  const int koffA = (lane >> 4) * 16;  // A: contiguous 16-group per lane-half
  const int koffB = (lane >> 4) * 16;  // B: contiguous 16-group per lane-half

  v8f zero = {};
  v8f acc[4] = {zero, zero, zero, zero};

  const float* xrow = x + (size_t)(row0 + m) * CIN;

  for (int kk = 0; kk < CIN; kk += 32) {
    // ---- A fragment: x row m, K = kk+koffA..+15
    v16h afrag;
    {
      _Float16* ad = (_Float16*)&afrag;
      const float4* p0 = (const float4*)(xrow + kk + koffA);
      const float4* p1 = (const float4*)(xrow + kk + koffA + 8);
      float4 f0 = p0[0], f1 = p0[1], f2 = p1[0], f3 = p1[1];
      ad[0]=(_Float16)f0.x; ad[1]=(_Float16)f0.y; ad[2]=(_Float16)f0.z; ad[3]=(_Float16)f0.w;
      ad[4]=(_Float16)f1.x; ad[5]=(_Float16)f1.y; ad[6]=(_Float16)f1.z; ad[7]=(_Float16)f1.w;
      ad[8]=(_Float16)f2.x; ad[9]=(_Float16)f2.y; ad[10]=(_Float16)f2.z; ad[11]=(_Float16)f2.w;
      ad[12]=(_Float16)f3.x; ad[13]=(_Float16)f3.y; ad[14]=(_Float16)f3.z; ad[15]=(_Float16)f3.w;
    }
#pragma unroll
    for (int t = 0; t < 4; ++t) {
      // ---- B fragment: W row o (col of B), 16 consecutive K values
      const int o = o0 + t * 16 + m;
      const float4* q = (const float4*)(W + (size_t)o * CIN + kk + koffB);
      float4 g0 = q[0], g1 = q[1], g2 = q[2], g3 = q[3];
      v16h bfrag;
      _Float16* bd = (_Float16*)&bfrag;
      bd[0]=(_Float16)g0.x; bd[1]=(_Float16)g0.y; bd[2]=(_Float16)g0.z; bd[3]=(_Float16)g0.w;
      bd[4]=(_Float16)g1.x; bd[5]=(_Float16)g1.y; bd[6]=(_Float16)g1.z; bd[7]=(_Float16)g1.w;
      bd[8]=(_Float16)g2.x; bd[9]=(_Float16)g2.y; bd[10]=(_Float16)g2.z; bd[11]=(_Float16)g2.w;
      bd[12]=(_Float16)g3.x; bd[13]=(_Float16)g3.y; bd[14]=(_Float16)g3.z; bd[15]=(_Float16)g3.w;

      acc[t] = __builtin_amdgcn_wmma_f32_16x16x32_f16(
          false, afrag, false, bfrag, (short)0, acc[t], false, false);
    }
  }

  // ---- store: C/D layout: VGPR r, lanes 0-15 -> M=r, lanes 16-31 -> M=r+8, N=lane&15
  const int lh = lane >> 4;
#pragma unroll
  for (int r = 0; r < 8; ++r) {
    const int row = row0 + r + 8 * lh;
    const int bb = row >> 10;
    const int n  = row & 1023;
#pragma unroll
    for (int t = 0; t < 4; ++t) {
      const int o = o0 + t * 16 + m;
      const float v = acc[t][r];
      hf32[(size_t)row * COUT + o] = v;
      const int hh = o >> 5, d = o & 31;
      hT[(((size_t)bb * HH + hh) * DD + d) * NN + n] = (_Float16)v;
    }
  }
}

// ---------------------------------------------------------------------------
// Kernel 2: per-node per-head attention scores  s_src / s_dst
// ---------------------------------------------------------------------------
__global__ __launch_bounds__(256) void gat_scores(
    const float* __restrict__ hf32, const float* __restrict__ a,
    float* __restrict__ ssrc, float* __restrict__ sdst) {
  const int t = blockIdx.x * 256 + threadIdx.x;   // 65536 = B*N*H
  const int hh = t & 7;
  const size_t bn = (size_t)(t >> 3);             // b*N + n
  const float* hr = hf32 + bn * COUT + hh * DD;
  const float* as = a + hh * (2 * DD);
  float s0 = 0.f, s1 = 0.f;
#pragma unroll 8
  for (int d = 0; d < DD; ++d) {
    const float v = hr[d];
    s0 += v * as[d];
    s1 += v * as[DD + d];
  }
  ssrc[bn * HH + hh] = s0;
  sdst[bn * HH + hh] = s1;
}

// ---------------------------------------------------------------------------
// Kernel 3: fused masked-softmax attention + aggregate + ELU.
// One block = (b, h, 16-row tile).  Pass1: row max.  Pass2: p=exp(e-max) -> LDS f16.
// Stage B: 8 waves K-split WMMA of p(16x1024) @ hT(1024x32), LDS reduce, scale, ELU.
// ---------------------------------------------------------------------------
__global__ __launch_bounds__(256) void gat_attn(
    const int* __restrict__ adj, const float* __restrict__ ssrc,
    const float* __restrict__ sdst, const _Float16* __restrict__ hT,
    float* __restrict__ out) {
  __shared__ _Float16 p_buf[16 * NN];       // 32 KB unnormalized softmax
  __shared__ float part[8 * 2 * 8 * 32];    // 16 KB wave partials
  __shared__ float sdst_sh[NN];             // 4 KB
  __shared__ float inv_sum_sh[16];

  const int b = blockIdx.z;
  const int h = blockIdx.y;
  const int i0 = blockIdx.x * 16;
  const int tid = threadIdx.x;

  for (int j = tid; j < NN; j += 256)
    sdst_sh[j] = sdst[((size_t)b * NN + j) * HH + h];
  __syncthreads();

  const int ii = tid >> 4;   // row 0..15
  const int r  = tid & 15;   // 16 threads per row
  const int i = i0 + ii;
  const int* adjrow = adj + ((size_t)b * NN + i) * NN;
  const float si = ssrc[((size_t)b * NN + i) * HH + h];

  // pass 1: row max
  float lmax = -3.0e38f;
  for (int j = r; j < NN; j += 16) {
    float ev = si + sdst_sh[j];
    ev = ev > 0.0f ? ev : 0.2f * ev;
    if (adjrow[j] == 0) ev = -1.0e9f;
    lmax = fmaxf(lmax, ev);
  }
#pragma unroll
  for (int off = 8; off >= 1; off >>= 1)
    lmax = fmaxf(lmax, __shfl_xor(lmax, off, 16));

  // pass 2: p = exp(e - max), sum
  float lsum = 0.f;
  for (int j = r; j < NN; j += 16) {
    float ev = si + sdst_sh[j];
    ev = ev > 0.0f ? ev : 0.2f * ev;
    if (adjrow[j] == 0) ev = -1.0e9f;
    const float p = __expf(ev - lmax);
    p_buf[ii * NN + j] = (_Float16)p;
    lsum += p;
  }
#pragma unroll
  for (int off = 8; off >= 1; off >>= 1)
    lsum += __shfl_xor(lsum, off, 16);
  if (r == 0) inv_sum_sh[ii] = 1.0f / lsum;
  __syncthreads();

  // stage B: wave w handles K in [w*128, w*128+128), both 16-col tiles of D=32
  const int w = tid >> 5, lane = tid & 31;
  const int m = lane & 15;
  const int koffA = (lane >> 4) * 16;
  const int koffB = (lane >> 4) * 16;
  const _Float16* hTb = hT + ((size_t)b * HH + h) * DD * NN;

  v8f acc0 = {}, acc1 = {};
#pragma unroll
  for (int kc = 0; kc < 4; ++kc) {
    const int kk = w * 128 + kc * 32;
    v16h afrag;
    {
      const v8h* pa = (const v8h*)(p_buf + m * NN + kk + koffA);
      ((v8h*)&afrag)[0] = pa[0];
      ((v8h*)&afrag)[1] = pa[1];   // +8 halves
    }
    v16h b0, b1;
    {
      const v8h* pb = (const v8h*)(hTb + (size_t)m * NN + kk + koffB);
      ((v8h*)&b0)[0] = pb[0]; ((v8h*)&b0)[1] = pb[1];
      const v8h* pc = (const v8h*)(hTb + (size_t)(m + 16) * NN + kk + koffB);
      ((v8h*)&b1)[0] = pc[0]; ((v8h*)&b1)[1] = pc[1];
    }
    acc0 = __builtin_amdgcn_wmma_f32_16x16x32_f16(false, afrag, false, b0, (short)0, acc0, false, false);
    acc1 = __builtin_amdgcn_wmma_f32_16x16x32_f16(false, afrag, false, b1, (short)0, acc1, false, false);
  }

  float* pw0 = part + ((w * 2 + 0) * 8) * 32;
  float* pw1 = part + ((w * 2 + 1) * 8) * 32;
#pragma unroll
  for (int rr = 0; rr < 8; ++rr) {
    pw0[rr * 32 + lane] = acc0[rr];
    pw1[rr * 32 + lane] = acc1[rr];
  }
  __syncthreads();

  // reduce 8 wave partials for 16x32 outputs, scale by 1/sum, ELU, store
  for (int idx = tid; idx < 512; idx += 256) {
    const int mm = idx >> 5;            // 0..15
    const int n  = idx & 31;            // 0..31
    const int t  = n >> 4;
    const int nn = n & 15;
    const int rr = mm & 7;
    const int ln = nn + 16 * (mm >> 3);
    float s = 0.f;
#pragma unroll
    for (int ww = 0; ww < 8; ++ww)
      s += part[((ww * 2 + t) * 8 + rr) * 32 + ln];
    s *= inv_sum_sh[mm];
    s = s > 0.0f ? s : (__expf(s) - 1.0f);   // ELU
    out[((size_t)b * NN + i0 + mm) * COUT + h * DD + n] = s;
  }
}

// ---------------------------------------------------------------------------
extern "C" void kernel_launch(void* const* d_in, const int* in_sizes, int n_in,
                              void* d_out, int out_size, void* d_ws, size_t ws_size,
                              hipStream_t stream) {
  const float* x   = (const float*)d_in[0];  // [8,1024,256]
  const int*   adj = (const int*)d_in[1];    // [8,1024,1024]
  const float* W   = (const float*)d_in[2];  // [256,256]
  const float* a   = (const float*)d_in[3];  // [1,8,64]
  float* out = (float*)d_out;                // [8,1024,256]

  char* ws = (char*)d_ws;
  float*    hf32 = (float*)(ws);                          // 8 MB
  _Float16* hT   = (_Float16*)(ws + (size_t)8 * 1024 * 1024);   // 4 MB
  float*    ssrc = (float*)(ws + (size_t)12 * 1024 * 1024);     // 256 KB
  float*    sdst = (float*)(ws + (size_t)12 * 1024 * 1024 + 256 * 1024);

  // GEMM1: 2048 wave-tiles / 8 waves per block
  gat_gemm1<<<256, 256, 0, stream>>>(x, W, hf32, hT);
  // scores: B*N*H = 65536 threads
  gat_scores<<<256, 256, 0, stream>>>(hf32, a, ssrc, sdst);
  // attention: (i_tile, h, b)
  gat_attn<<<dim3(NN / 16, HH, BQ), 256, 0, stream>>>(adj, ssrc, sdst, hT, out);
}
